// CausalAttention_52682068853258
// MI455X (gfx1250) — matrix operand order repaired and checked by reading, compiled, not run on hardware
//
#include <hip/hip_runtime.h>
#include <hip/hip_bf16.h>
#include <cstdint>

// ---------------------------------------------------------------------------
// Causal MHA for MI455X (gfx1250, wave32, WMMA 16x16x32 f16->f32)
// B=4, S=2048, D=2048, H=16, Dh=128
// ---------------------------------------------------------------------------

#define BATCH 4
#define SEQ   2048
#define DMODEL 2048
#define NHEADS 16
#define DHEAD 128
#define MTOT  (BATCH * SEQ)            // 8192

typedef _Float16 h8   __attribute__((ext_vector_type(8)));
typedef _Float16 v16h __attribute__((ext_vector_type(16)));
typedef float    v8f  __attribute__((ext_vector_type(8)));

// ---- WMMA fragment loaders (CDNA5 ISA 7.12.2 layouts) ----------------------
// A-matrix 16x32 f16: lane l holds row (l&15); lanes<16: K 0..7 & 16..23,
// lanes>=16: K 8..15 & 24..31  (two 16B loads)
__device__ __forceinline__ v16h load_frag_a(const _Float16* base, int stride, int lane) {
    int r  = lane & 15;
    int kb = (lane >> 4) << 3;          // 0 or 8
    const _Float16* p = base + r * stride + kb;
    h8 lo = *(const h8*)(p);
    h8 hi = *(const h8*)(p + 16);
    return __builtin_shufflevector(lo, hi, 0,1,2,3,4,5,6,7,8,9,10,11,12,13,14,15);
}
// B-matrix 32x16 f16 (stored as row-major [N][K] slab): lane l holds col (l&15),
// lanes<16: K 0..7 & 16..23, lanes>=16: K 8..15 & 24..31 (two 16B loads)
__device__ __forceinline__ v16h load_frag_b(const _Float16* base, int stride, int lane) {
    int n  = lane & 15;
    int kb = (lane >> 4) << 3;          // 0 or 8
    const _Float16* p = base + n * stride + kb;
    h8 lo = *(const h8*)(p);
    h8 hi = *(const h8*)(p + 16);
    return __builtin_shufflevector(lo, hi, 0,1,2,3,4,5,6,7,8,9,10,11,12,13,14,15);
}

__device__ __forceinline__ v8f wmma_f16(v16h a, v16h b, v8f c) {
    return __builtin_amdgcn_wmma_f32_16x16x32_f16(false, a, false, b, (short)0, c, false, false);
}

// ---- CDNA5 async global->LDS copy (ASYNCcnt-tracked, no VGPR round-trip) ---
// Per lane: LDS[lds_off] = MEM[gaddr] (16 bytes).
__device__ __forceinline__ void async_copy_b128(uint32_t lds_off, const void* gptr) {
    asm volatile("global_load_async_to_lds_b128 %0, %1, off"
                 :: "v"(lds_off), "v"(gptr)
                 : "memory");
}
__device__ __forceinline__ void wait_async0() {
    asm volatile("s_wait_asynccnt 0" ::: "memory");
}
// Generic LDS pointer -> 32-bit LDS byte address (low 32 bits of aperture addr)
__device__ __forceinline__ uint32_t lds_addr32(const void* p) {
    return (uint32_t)(uintptr_t)p;
}

// ---------------------------------------------------------------------------
// fp32 -> fp16 conversion
// ---------------------------------------------------------------------------
__global__ void cvt_f32_f16_kernel(const float* __restrict__ src,
                                   _Float16* __restrict__ dst, int n) {
    for (int i = blockIdx.x * blockDim.x + threadIdx.x; i < n;
         i += gridDim.x * blockDim.x)
        dst[i] = (_Float16)src[i];
}

// ---------------------------------------------------------------------------
// Per-head transpose: v16[B,S,H,Dh] -> vt[B,H,Dh,S]
// ---------------------------------------------------------------------------
__global__ void transpose_v_kernel(const _Float16* __restrict__ v16,
                                   _Float16* __restrict__ vt) {
    int total = BATCH * NHEADS * DHEAD * SEQ;
    for (int i = blockIdx.x * blockDim.x + threadIdx.x; i < total;
         i += gridDim.x * blockDim.x) {
        int s = i & (SEQ - 1);
        int t = i >> 11;
        int d = t & (DHEAD - 1);
        t >>= 7;
        int h = t & (NHEADS - 1);
        int b = t >> 4;
        vt[i] = v16[((size_t)(b * SEQ + s)) * DMODEL + h * DHEAD + d];
    }
}

// ---------------------------------------------------------------------------
// NT GEMM: C[M,N] = A[M,K] * W[N,K]^T.  128x128 block tile, K-step 32.
// Double-buffered LDS, tiles staged with global_load_async_to_lds_b128.
// 8 waves: 4(M) x 2(N), each wave 32x64 = 8 WMMA frags per K-step.
// ---------------------------------------------------------------------------
#define LDS_STRIDE 40  // 32 + 8 halves pad; 80B rows keep 16B alignment

template <bool F32OUT>
__global__ __launch_bounds__(256)
void gemm_nt_kernel(const _Float16* __restrict__ A,
                    const _Float16* __restrict__ W,
                    _Float16* __restrict__ C16,
                    float* __restrict__ C32,
                    const float* __restrict__ bias,
                    int M, int N, int K) {
    __shared__ _Float16 As[2][128 * LDS_STRIDE];
    __shared__ _Float16 Bs[2][128 * LDS_STRIDE];

    const int tid  = threadIdx.x;
    const int lane = tid & 31;
    const int wave = tid >> 5;
    const int wy   = wave & 3;   // 0..3  (M groups of 32)
    const int wx   = wave >> 2;  // 0..1  (N groups of 64)

    const int m0 = blockIdx.y * 128;
    const int n0 = blockIdx.x * 128;

    // Per-thread staging coordinates: 2 x 16B for A tile, 2 x 16B for B tile.
    int row0 = (tid * 8) >> 5;           // pass 0: rows 0..63
    int col0 = (tid * 8) & 31;
    int row1 = row0 + 64;                // pass 1: rows 64..127
    int col1 = col0;

    auto stage = [&](int buf, int kt) {
        async_copy_b128(lds_addr32(&As[buf][row0 * LDS_STRIDE + col0]),
                        A + (size_t)(m0 + row0) * K + kt + col0);
        async_copy_b128(lds_addr32(&As[buf][row1 * LDS_STRIDE + col1]),
                        A + (size_t)(m0 + row1) * K + kt + col1);
        async_copy_b128(lds_addr32(&Bs[buf][row0 * LDS_STRIDE + col0]),
                        W + (size_t)(n0 + row0) * K + kt + col0);
        async_copy_b128(lds_addr32(&Bs[buf][row1 * LDS_STRIDE + col1]),
                        W + (size_t)(n0 + row1) * K + kt + col1);
    };

    v8f acc[2][4];
#pragma unroll
    for (int i = 0; i < 2; ++i)
#pragma unroll
        for (int j = 0; j < 4; ++j) acc[i][j] = (v8f)0.0f;

    const int nk = K / 32;
    stage(0, 0);

    for (int it = 0; it < nk; ++it) {
        wait_async0();       // this wave's async copies into buf landed
        __syncthreads();     // all waves' copies landed; prev compute done
        if (it + 1 < nk) stage((it + 1) & 1, (it + 1) * 32);

        const int buf = it & 1;
        v16h afrag[2], bfrag[4];
#pragma unroll
        for (int i = 0; i < 2; ++i)
            afrag[i] = load_frag_a(&As[buf][(wy * 32 + i * 16) * LDS_STRIDE],
                                   LDS_STRIDE, lane);
#pragma unroll
        for (int j = 0; j < 4; ++j)
            bfrag[j] = load_frag_b(&Bs[buf][(wx * 64 + j * 16) * LDS_STRIDE],
                                   LDS_STRIDE, lane);

#pragma unroll
        for (int i = 0; i < 2; ++i)
#pragma unroll
            for (int j = 0; j < 4; ++j)
                acc[i][j] = wmma_f16(afrag[i], bfrag[j], acc[i][j]);
    }

    // epilogue: C frag layout -> lane holds (row = v + 8*(l>=16), col = l&15)
    const int hl = lane >> 4;
    const int cn = lane & 15;
#pragma unroll
    for (int i = 0; i < 2; ++i)
#pragma unroll
        for (int j = 0; j < 4; ++j)
#pragma unroll
            for (int v = 0; v < 8; ++v) {
                int row = m0 + wy * 32 + i * 16 + v + 8 * hl;
                int col = n0 + wx * 64 + j * 16 + cn;
                if (F32OUT)
                    C32[(size_t)row * N + col] = acc[i][j][v] + bias[col];
                else
                    C16[(size_t)row * N + col] = (_Float16)acc[i][j][v];
            }
}

// ---------------------------------------------------------------------------
// Flash-style causal attention.  One wave per 16-row Q tile.
// q16,k16: [B,S,D] (head slices contiguous); vt: [B,H,Dh,S]; o16: [B,S,D].
// ---------------------------------------------------------------------------
#define ATT_WAVES 8
#define P_STRIDE  40

__global__ __launch_bounds__(256)
void attention_kernel(const _Float16* __restrict__ q16,
                      const _Float16* __restrict__ k16,
                      const _Float16* __restrict__ vt,
                      _Float16* __restrict__ o16) {
    __shared__ _Float16 Plds[ATT_WAVES * 16 * P_STRIDE];

    const int lane = threadIdx.x & 31;
    const int wave = threadIdx.x >> 5;
    const int tile = blockIdx.x * ATT_WAVES + wave;   // 0 .. B*H*S/16-1

    const int qtile = tile & (SEQ / 16 - 1);
    const int bh    = tile >> 7;                      // S/16 = 128
    const int h     = bh & (NHEADS - 1);
    const int b     = bh >> 4;
    const int q0    = qtile * 16;

    const int hl = lane >> 4;
    const int cn = lane & 15;
    const float sm_scale = 0.08838834764831845f;      // 1/sqrt(128)

    _Float16* myP = &Plds[wave * 16 * P_STRIDE];

    const _Float16* qbase = q16 + (size_t)(b * SEQ + q0) * DMODEL + h * DHEAD;
    const _Float16* kbase = k16 + (size_t)(b * SEQ) * DMODEL + h * DHEAD;
    const _Float16* vtb   = vt + (size_t)bh * DHEAD * SEQ;

    // Q fragments for this tile: 4 chunks of K=32 covering Dh=128
    v16h qfrag[4];
#pragma unroll
    for (int c = 0; c < 4; ++c)
        qfrag[c] = load_frag_a(qbase + c * 32, DMODEL, lane);

    v8f oacc[8];
#pragma unroll
    for (int j = 0; j < 8; ++j) oacc[j] = (v8f)0.0f;
    float mrow[8], lrow[8];
#pragma unroll
    for (int v = 0; v < 8; ++v) { mrow[v] = -1e30f; lrow[v] = 0.0f; }

    const int kend = q0 + 15;                         // last causal column
    for (int kc = 0; kc <= kend; kc += 32) {
        // prime next chunk's K rows and Vt rows (global_prefetch_b8)
        if (kc + 32 <= kend) {
            __builtin_prefetch(kbase + (size_t)(kc + 32 + (lane & 31)) * DMODEL, 0, 1);
            __builtin_prefetch(vtb + (size_t)(lane * 4) * SEQ + kc + 32, 0, 1);
        }

        // ---- scores: two 16x16 sub-tiles over 32 k-positions ----
        v8f sfr[2];
        sfr[0] = (v8f)0.0f; sfr[1] = (v8f)0.0f;
#pragma unroll
        for (int t = 0; t < 2; ++t)
#pragma unroll
            for (int c = 0; c < 4; ++c) {
                v16h kb = load_frag_b(kbase + (size_t)(kc + t * 16) * DMODEL + c * 32,
                                      DMODEL, lane);
                sfr[t] = wmma_f16(qfrag[c], kb, sfr[t]);
            }

        // ---- scale + causal mask ----
        float s0[8], s1[8];
#pragma unroll
        for (int v = 0; v < 8; ++v) {
            int qrow = q0 + v + 8 * hl;
            float a = sfr[0][v] * sm_scale;
            float c2 = sfr[1][v] * sm_scale;
            if (kc + cn > qrow)      a  = -1e30f;
            if (kc + 16 + cn > qrow) c2 = -1e30f;
            s0[v] = a; s1[v] = c2;
        }

        // ---- online softmax: row max over 32 cols (shfl within 16-group) ----
        float smax[8];
#pragma unroll
        for (int v = 0; v < 8; ++v) {
            float mx = fmaxf(s0[v], s1[v]);
#pragma unroll
            for (int m = 1; m < 16; m <<= 1)
                mx = fmaxf(mx, __shfl_xor(mx, m, 32));
            smax[v] = mx;
        }

        float corr[8], p0[8], p1[8];
#pragma unroll
        for (int v = 0; v < 8; ++v) {
            float nm = fmaxf(mrow[v], smax[v]);
            corr[v]  = __expf(mrow[v] - nm);
            p0[v]    = __expf(s0[v] - nm);
            p1[v]    = __expf(s1[v] - nm);
            float rs = p0[v] + p1[v];
#pragma unroll
            for (int m = 1; m < 16; m <<= 1)
                rs += __shfl_xor(rs, m, 32);
            mrow[v]  = nm;
            lrow[v]  = lrow[v] * corr[v] + rs;
        }

        // ---- rescale O accumulators ----
#pragma unroll
        for (int j = 0; j < 8; ++j)
#pragma unroll
            for (int v = 0; v < 8; ++v) oacc[j][v] *= corr[v];

        // ---- bounce P through LDS: C-frag layout -> A-frag layout ----
#pragma unroll
        for (int v = 0; v < 8; ++v) {
            myP[(v + 8 * hl) * P_STRIDE + cn]      = (_Float16)p0[v];
            myP[(v + 8 * hl) * P_STRIDE + 16 + cn] = (_Float16)p1[v];
        }
        v16h pfrag = load_frag_a(myP, P_STRIDE, lane);

        // ---- O += P * V  (Vt rows are d, contiguous in k) ----
#pragma unroll
        for (int j = 0; j < 8; ++j) {
            v16h vb = load_frag_b(vtb + (size_t)(j * 16) * SEQ + kc, SEQ, lane);
            oacc[j] = wmma_f16(pfrag, vb, oacc[j]);
        }
    }

    // ---- normalize and store to o16 [B,S,D] ----
    _Float16* obase = o16 + (size_t)(b * SEQ + q0) * DMODEL + h * DHEAD;
#pragma unroll
    for (int j = 0; j < 8; ++j)
#pragma unroll
        for (int v = 0; v < 8; ++v) {
            int row = v + 8 * hl;
            obase[(size_t)row * DMODEL + j * 16 + cn] =
                (_Float16)(oacc[j][v] / lrow[v]);
        }
}

// ---------------------------------------------------------------------------
// Host launcher
// ---------------------------------------------------------------------------
extern "C" void kernel_launch(void* const* d_in, const int* in_sizes, int n_in,
                              void* d_out, int out_size, void* d_ws, size_t ws_size,
                              hipStream_t stream) {
    const float* x  = (const float*)d_in[0];
    const float* Wq = (const float*)d_in[1];
    const float* Wk = (const float*)d_in[2];
    const float* Wv = (const float*)d_in[3];
    const float* Wo = (const float*)d_in[4];
    const float* bo = (const float*)d_in[5];
    float* out = (float*)d_out;

    const size_t MB = 1024ull * 1024ull;
    char* ws = (char*)d_ws;
    _Float16* x16  = (_Float16*)(ws + 0 * MB);
    _Float16* wq16 = (_Float16*)(ws + 32 * MB);
    _Float16* wk16 = (_Float16*)(ws + 40 * MB);
    _Float16* wv16 = (_Float16*)(ws + 48 * MB);
    _Float16* wo16 = (_Float16*)(ws + 56 * MB);
    _Float16* q16  = (_Float16*)(ws + 64 * MB);
    _Float16* k16  = (_Float16*)(ws + 96 * MB);
    _Float16* v16  = (_Float16*)(ws + 128 * MB);
    _Float16* vtb  = (_Float16*)(ws + 160 * MB);
    _Float16* o16  = v16;  // reuse V buffer after transpose

    const int nX = MTOT * DMODEL;
    const int nW = DMODEL * DMODEL;
    cvt_f32_f16_kernel<<<2048, 256, 0, stream>>>(x,  x16,  nX);
    cvt_f32_f16_kernel<<<1024, 256, 0, stream>>>(Wq, wq16, nW);
    cvt_f32_f16_kernel<<<1024, 256, 0, stream>>>(Wk, wk16, nW);
    cvt_f32_f16_kernel<<<1024, 256, 0, stream>>>(Wv, wv16, nW);
    cvt_f32_f16_kernel<<<1024, 256, 0, stream>>>(Wo, wo16, nW);

    dim3 ggrid(DMODEL / 128, MTOT / 128);  // (16, 64)
    gemm_nt_kernel<false><<<ggrid, 256, 0, stream>>>(x16, wq16, q16, nullptr, nullptr,
                                                     MTOT, DMODEL, DMODEL);
    gemm_nt_kernel<false><<<ggrid, 256, 0, stream>>>(x16, wk16, k16, nullptr, nullptr,
                                                     MTOT, DMODEL, DMODEL);
    gemm_nt_kernel<false><<<ggrid, 256, 0, stream>>>(x16, wv16, v16, nullptr, nullptr,
                                                     MTOT, DMODEL, DMODEL);

    transpose_v_kernel<<<4096, 256, 0, stream>>>(v16, vtb);

    int ntiles = BATCH * NHEADS * (SEQ / 16);          // 8192
    attention_kernel<<<ntiles / ATT_WAVES, 256, 0, stream>>>(q16, k16, vtb, o16);

    gemm_nt_kernel<true><<<ggrid, 256, 0, stream>>>(o16, wo16, nullptr, out, bo,
                                                    MTOT, DMODEL, DMODEL);
}
